// trajectory2seq_6665789243665
// MI455X (gfx1250) — matrix operand order repaired and checked by reading, not compiled, not run on hardware
//
#include <hip/hip_runtime.h>
#include <cstdint>

#define DEV __device__ __forceinline__

typedef __attribute__((ext_vector_type(16))) _Float16 v16h;
typedef __attribute__((ext_vector_type(8)))  float    v8f;

constexpr int H = 20, G = 60, DIN = 64, S = 128, VOCAB = 128, MAXDEC = 6;
constexpr int Bsz = 8192;
constexpr int HP = 32;   // hidden padded to one K-tile
constexpr int GP = 64;   // gates (3H=60) padded to 4 N-tiles

// ---- CDNA5 async Global->LDS path (ASYNCcnt), with sync fallback ----------
#if defined(__has_builtin)
#if __has_builtin(__builtin_amdgcn_global_load_async_to_lds_b128) && \
    __has_builtin(__builtin_amdgcn_s_wait_asynccnt)
#define HAVE_ASYNC_LDS 1
#endif
#endif

#ifdef HAVE_ASYNC_LDS
#define AS1 __attribute__((address_space(1)))
#define AS3 __attribute__((address_space(3)))
typedef int v4i_vs __attribute__((vector_size(4 * sizeof(int))));
DEV void async_ld_b128(void* lds_generic, const void* gaddr) {
  // generic LDS pointer low 32 bits == LDS byte address (aperture mapping);
  // generic global address == global address. Integer casts -> addrspacecast.
  AS3 v4i_vs* l = (AS3 v4i_vs*)(uint32_t)(uintptr_t)lds_generic;
  AS1 v4i_vs* g = (AS1 v4i_vs*)(uintptr_t)gaddr;
  __builtin_amdgcn_global_load_async_to_lds_b128(g, l, 0, 0);
}
#endif

DEV v8f wmma16(v16h a, v16h b, v8f c) {
  return __builtin_amdgcn_wmma_f32_16x16x32_f16(false, a, false, b, (short)0, c,
                                                false, false);
}

DEV v8f splat8(float v) {
  v8f c;
#pragma unroll
  for (int i = 0; i < 8; ++i) c[i] = v;
  return c;
}

// A-matrix fragment (16xK f16), source row-major fp32 [16 x ld], K-tile kt.
// Layout (ISA 7.12.2): lane m=lane&15, half=lane>>4; VGPR r<4 -> K=8*half+e,
// VGPR r>=4 -> K=16+8*half+(e-8).
DEV v16h a_frag(const float* src, int ld, int kt, int kvalid, int lane) {
  int m = lane & 15, half = lane >> 4;
  v16h a;
#pragma unroll
  for (int e = 0; e < 16; ++e) {
    int k = kt * 32 + ((e < 8) ? (half * 8 + e) : (8 + half * 8 + e));
    float v = (k < kvalid) ? src[m * ld + k] : 0.0f;
    a[e] = (_Float16)v;
  }
  return a;
}

// B-matrix fragment (32x16 f16) of W^T where W is [nrows x ldw] row-major fp32.
// B[k][n] = W[n][k]; lane n=lane&15, half=lane>>4, element e<8 -> K=32*kt+8*half+e, e>=8 -> K=32*kt+16+8*half+(e-8).
DEV v16h b_fragT(const float* W, int ldw, int nrows, int kvalid, int nt, int kt,
                 int lane) {
  int n = nt * 16 + (lane & 15);
  int half = lane >> 4;
  v16h b;
#pragma unroll
  for (int e = 0; e < 16; ++e) {
    int k = kt * 32 + ((e < 8) ? (half * 8 + e) : (8 + half * 8 + e));
    float v = (n < nrows && k < kvalid) ? W[n * ldw + k] : 0.0f;
    b[e] = (_Float16)v;
  }
  return b;
}

// D tile (16x16 f32) -> LDS row-major [16 x ldd]; M = r + 8*half, N = lane&15.
DEV void store_d(float* dst, int ldd, int nt, v8f d, int lane) {
  int n = nt * 16 + (lane & 15), half = lane >> 4;
#pragma unroll
  for (int r = 0; r < 8; ++r) dst[(r + half * 8) * ldd + n] = d[r];
}

// GRU gate combine: gi/gh [16 x GP] (bias already folded), h [16 x HP] in/out.
DEV void gru_update(const float* gi, const float* gh, float* h, int lane) {
  int m = lane & 15, half = lane >> 4;
#pragma unroll
  for (int kk = 0; kk < 10; ++kk) {
    int k = half * 10 + kk;
    float r = 1.0f / (1.0f + __expf(-(gi[m * GP + k] + gh[m * GP + k])));
    float z = 1.0f / (1.0f + __expf(-(gi[m * GP + 20 + k] + gh[m * GP + 20 + k])));
    float n = tanhf(gi[m * GP + 40 + k] + r * gh[m * GP + 40 + k]);
    h[m * HP + k] = (1.0f - z) * n + z * h[m * HP + k];
  }
}

DEV float dot20(const float* a, const float* b) {
  const float4* a4 = (const float4*)a;
  const float4* b4 = (const float4*)b;
  float s = 0.f;
#pragma unroll
  for (int i = 0; i < 5; ++i) {
    float4 x = a4[i], y = b4[i];
    s += x.x * y.x;
    s += x.y * y.y;
    s += x.z * y.z;
    s += x.w * y.w;
  }
  return s;
}

// Issue (async) or perform (sync) the 16x64 fp32 x-tile copy into LDS.
DEV void xtile_async_issue(const float* x, int rowBase, int t, float* dstLds,
                           int lane) {
#ifdef HAVE_ASYNC_LDS
#pragma unroll
  for (int j = 0; j < 8; ++j) {
    int idx = lane + 32 * j;  // 0..255 float4s; dst byte off = idx*16
    int row = idx >> 4, c4 = idx & 15;
    const float* g = x + (((size_t)(rowBase + row)) * S + t) * DIN + c4 * 4;
    async_ld_b128(dstLds + idx * 4, g);
  }
#else
#pragma unroll
  for (int j = 0; j < 8; ++j) {
    int idx = lane + 32 * j;
    int row = idx >> 4, c4 = idx & 15;
    const float4* g =
        (const float4*)(x + (((size_t)(rowBase + row)) * S + t) * DIN) + c4;
    ((float4*)dstLds)[idx] = *g;
  }
#endif
}

// ---------------------------------------------------------------------------
// Encoder: one wave = 16 batch rows, 128 sequential 2-layer GRU steps.
// Weights held in VGPRs as f16 B-fragments; gate tiles via WMMA; nonlinearity
// through per-wave LDS. x-tile double-buffered in LDS with async Global->LDS
// prefetch of step t+1 overlapped with step t's compute.
// ---------------------------------------------------------------------------
__global__ __launch_bounds__(64) void t2s_encoder(
    const float* __restrict__ x, const float* __restrict__ Wih0,
    const float* __restrict__ Whh0, const float* __restrict__ bih0,
    const float* __restrict__ bhh0, const float* __restrict__ Wih1,
    const float* __restrict__ Whh1, const float* __restrict__ bih1,
    const float* __restrict__ bhh1, float* __restrict__ encoded,
    float* __restrict__ h0_out, float* __restrict__ h1_out) {
  __shared__ __align__(16) float smem[2 * 5120];
  int lane = threadIdx.x & 31, wv = threadIdx.x >> 5;
  float* xbuf = smem + wv * 5120;  // 2 x [16 x 64] double buffer
  float* gi = xbuf + 2048;         // [16 x GP]
  float* gh = xbuf + 3072;         // [16 x GP]
  float* h0 = xbuf + 4096;         // [16 x HP]
  float* h1 = xbuf + 4608;         // [16 x HP]
  int rowBase = (blockIdx.x * 2 + wv) * 16;

  for (int i = lane; i < 512; i += 32) { h0[i] = 0.f; h1[i] = 0.f; }

  // prefetch tile 0 while we gather weight fragments
  xtile_async_issue(x, rowBase, 0, xbuf, lane);

  v16h fwih0[2][4], fwhh0[4], fwih1[4], fwhh1[4];
#pragma unroll
  for (int nt = 0; nt < 4; ++nt) {
    fwih0[0][nt] = b_fragT(Wih0, DIN, G, DIN, nt, 0, lane);
    fwih0[1][nt] = b_fragT(Wih0, DIN, G, DIN, nt, 1, lane);
    fwhh0[nt] = b_fragT(Whh0, H, G, H, nt, 0, lane);
    fwih1[nt] = b_fragT(Wih1, H, G, H, nt, 0, lane);
    fwhh1[nt] = b_fragT(Whh1, H, G, H, nt, 0, lane);
  }
  float cbi0[4], cbh0[4], cbi1[4], cbh1[4];
#pragma unroll
  for (int nt = 0; nt < 4; ++nt) {
    int g = nt * 16 + (lane & 15);
    cbi0[nt] = (g < G) ? bih0[g] : 0.f;
    cbh0[nt] = (g < G) ? bhh0[g] : 0.f;
    cbi1[nt] = (g < G) ? bih1[g] : 0.f;
    cbh1[nt] = (g < G) ? bhh1[g] : 0.f;
  }
  __syncthreads();

  for (int t = 0; t < S; ++t) {
    float* xt = xbuf + (t & 1) * 1024;
    // kick off next tile, then wait only for the *current* tile's 8 async
    // loads (async loads complete in order -> wait asynccnt<=8 drains tile t).
#ifdef HAVE_ASYNC_LDS
    if (t + 1 < S) {
      xtile_async_issue(x, rowBase, t + 1, xbuf + ((t + 1) & 1) * 1024, lane);
      __builtin_amdgcn_s_wait_asynccnt(8);
    } else {
      __builtin_amdgcn_s_wait_asynccnt(0);
    }
#else
    if (t + 1 < S)
      xtile_async_issue(x, rowBase, t + 1, xbuf + ((t + 1) & 1) * 1024, lane);
#endif
    __syncthreads();

    // layer 0
    v16h xa0 = a_frag(xt, 64, 0, DIN, lane);
    v16h xa1 = a_frag(xt, 64, 1, DIN, lane);
    v16h ha = a_frag(h0, HP, 0, H, lane);
#pragma unroll
    for (int nt = 0; nt < 4; ++nt) {
      v8f ci = splat8(cbi0[nt]);
      ci = wmma16(xa0, fwih0[0][nt], ci);
      ci = wmma16(xa1, fwih0[1][nt], ci);
      store_d(gi, GP, nt, ci, lane);
      v8f ch = splat8(cbh0[nt]);
      ch = wmma16(ha, fwhh0[nt], ch);
      store_d(gh, GP, nt, ch, lane);
    }
    __syncthreads();
    gru_update(gi, gh, h0, lane);
    __syncthreads();

    // layer 1
    v16h h0a = a_frag(h0, HP, 0, H, lane);
    v16h h1a = a_frag(h1, HP, 0, H, lane);
#pragma unroll
    for (int nt = 0; nt < 4; ++nt) {
      v8f ci = splat8(cbi1[nt]);
      ci = wmma16(h0a, fwih1[nt], ci);
      store_d(gi, GP, nt, ci, lane);
      v8f ch = splat8(cbh1[nt]);
      ch = wmma16(h1a, fwhh1[nt], ch);
      store_d(gh, GP, nt, ch, lane);
    }
    __syncthreads();
    gru_update(gi, gh, h1, lane);
    __syncthreads();

    {  // encoded[b,t,:] = h1
      int m = lane & 15, half = lane >> 4;
      float* dst = encoded + (((size_t)(rowBase + m)) * S + t) * H;
#pragma unroll
      for (int kk = 0; kk < 10; ++kk) {
        int k = half * 10 + kk;
        dst[k] = h1[m * HP + k];
      }
    }
  }
  {
    int m = lane & 15, half = lane >> 4;
#pragma unroll
    for (int kk = 0; kk < 10; ++kk) {
      int k = half * 10 + kk;
      h0_out[(size_t)(rowBase + m) * H + k] = h0[m * HP + k];
      h1_out[(size_t)(rowBase + m) * H + k] = h1[m * HP + k];
    }
  }
}

// ---------------------------------------------------------------------------
// Decoder: one wave = 16 batch rows, 6 autoregressive steps. WMMA for the
// GRU cells and the [16x20]@[20x128] logits matmul; VALU float4 dots for the
// per-row attention against L2-resident `encoded`; wave32 shfl reductions
// for softmax and argmax; coalesced float4 stores for logits/attention.
// ---------------------------------------------------------------------------
__global__ __launch_bounds__(64) void t2s_decoder(
    const float* __restrict__ dWih0, const float* __restrict__ dWhh0,
    const float* __restrict__ dbih0, const float* __restrict__ dbhh0,
    const float* __restrict__ dWih1, const float* __restrict__ dWhh1,
    const float* __restrict__ dbih1, const float* __restrict__ dbhh1,
    const float* __restrict__ emb, const float* __restrict__ qW,
    const float* __restrict__ qb, const float* __restrict__ combW,
    const float* __restrict__ combb, const float* __restrict__ outW,
    const float* __restrict__ outb, const float* __restrict__ encoded,
    const float* __restrict__ h0_in, const float* __restrict__ h1_in,
    float* __restrict__ vec_out, float* __restrict__ hidden_out,
    float* __restrict__ attn_out) {
  __shared__ __align__(16) float smem[2 * 7168];
  __shared__ int vecin_s[2][16];
  int lane = threadIdx.x & 31, wv = threadIdx.x >> 5;
  float* base = smem + wv * 7168;
  float* e = base;            // [16 x HP]
  float* d0 = base + 512;     // [16 x HP]
  float* d1 = base + 1024;    // [16 x HP]
  float* gi = base + 1536;    // [16 x GP]
  float* gh = base + 2560;    // [16 x GP]
  float* qry = base + 3584;   // [16 x HP]
  float* actx = base + 4096;  // [16 x HP]
  float* comb = base + 4608;  // [16 x HP]
  float* big = base + 5120;   // [16 x 128] scores / logits
  int* vin = vecin_s[wv];
  int rowBase = (blockIdx.x * 2 + wv) * 16;
  int m = lane & 15, half = lane >> 4;

  for (int i = lane; i < 5120; i += 32) base[i] = 0.f;
  if (lane < 16) vin[lane] = 0;

  v16h fwih0[4], fwhh0[4], fwih1[4], fwhh1[4], fout[8];
#pragma unroll
  for (int nt = 0; nt < 4; ++nt) {
    fwih0[nt] = b_fragT(dWih0, H, G, H, nt, 0, lane);
    fwhh0[nt] = b_fragT(dWhh0, H, G, H, nt, 0, lane);
    fwih1[nt] = b_fragT(dWih1, H, G, H, nt, 0, lane);
    fwhh1[nt] = b_fragT(dWhh1, H, G, H, nt, 0, lane);
  }
#pragma unroll
  for (int nt = 0; nt < 8; ++nt)
    fout[nt] = b_fragT(outW, H, VOCAB, H, nt, 0, lane);

  float cbi0[4], cbh0[4], cbi1[4], cbh1[4], cbo[8];
#pragma unroll
  for (int nt = 0; nt < 4; ++nt) {
    int g = nt * 16 + m;
    cbi0[nt] = (g < G) ? dbih0[g] : 0.f;
    cbh0[nt] = (g < G) ? dbhh0[g] : 0.f;
    cbi1[nt] = (g < G) ? dbih1[g] : 0.f;
    cbh1[nt] = (g < G) ? dbhh1[g] : 0.f;
  }
#pragma unroll
  for (int nt = 0; nt < 8; ++nt) cbo[nt] = outb[nt * 16 + m];

  __syncthreads();
  {  // initial hidden from encoder
#pragma unroll
    for (int kk = 0; kk < 10; ++kk) {
      int k = half * 10 + kk;
      d0[m * HP + k] = h0_in[(size_t)(rowBase + m) * H + k];
      d1[m * HP + k] = h1_in[(size_t)(rowBase + m) * H + k];
    }
  }
  __syncthreads();

  for (int step = 0; step < MAXDEC; ++step) {
    {  // embedding lookup
      int tok = vin[m];
      const float* ep = emb + (size_t)tok * H;
#pragma unroll
      for (int kk = 0; kk < 10; ++kk) {
        int k = half * 10 + kk;
        e[m * HP + k] = ep[k];
      }
    }
    __syncthreads();

    // GRU cell 0
    {
      v16h ea = a_frag(e, HP, 0, H, lane);
      v16h ha = a_frag(d0, HP, 0, H, lane);
#pragma unroll
      for (int nt = 0; nt < 4; ++nt) {
        v8f ci = splat8(cbi0[nt]);
        ci = wmma16(ea, fwih0[nt], ci);
        store_d(gi, GP, nt, ci, lane);
        v8f ch = splat8(cbh0[nt]);
        ch = wmma16(ha, fwhh0[nt], ch);
        store_d(gh, GP, nt, ch, lane);
      }
    }
    __syncthreads();
    gru_update(gi, gh, d0, lane);
    __syncthreads();

    // GRU cell 1
    {
      v16h d0a = a_frag(d0, HP, 0, H, lane);
      v16h d1a = a_frag(d1, HP, 0, H, lane);
#pragma unroll
      for (int nt = 0; nt < 4; ++nt) {
        v8f ci = splat8(cbi1[nt]);
        ci = wmma16(d0a, fwih1[nt], ci);
        store_d(gi, GP, nt, ci, lane);
        v8f ch = splat8(cbh1[nt]);
        ch = wmma16(d1a, fwhh1[nt], ch);
        store_d(gh, GP, nt, ch, lane);
      }
    }
    __syncthreads();
    gru_update(gi, gh, d1, lane);
    __syncthreads();

    // query = d1 @ qW^T + qb  (tiny: VALU)
#pragma unroll
    for (int kk = 0; kk < 10; ++kk) {
      int k = half * 10 + kk;
      float acc = qb[k];
      for (int j = 0; j < H; ++j) acc += d1[m * HP + j] * qW[k * H + j];
      qry[m * HP + k] = acc;
    }
    __syncthreads();

    // attention over S=128 (lane pair (m, m+16) splits the s-range)
    const float* enc_row0 = encoded + ((size_t)(rowBase + m) * S + half * 64) * H;
    const float* qp = qry + m * HP;
    float mx = -3.4e38f;
    for (int s = 0; s < 64; ++s) {
      float d = dot20(qp, enc_row0 + s * H);
      big[m * VOCAB + half * 64 + s] = d;
      mx = fmaxf(mx, d);
    }
    mx = fmaxf(mx, __shfl_xor(mx, 16, 32));
    float sum = 0.f;
    for (int s = 0; s < 64; ++s) {
      float w = __expf(big[m * VOCAB + half * 64 + s] - mx);
      big[m * VOCAB + half * 64 + s] = w;
      sum += w;
    }
    sum += __shfl_xor(sum, 16, 32);
    float inv = 1.0f / sum;
    float acc[20];
#pragma unroll
    for (int j = 0; j < 20; ++j) acc[j] = 0.f;
    float* attn_row =
        attn_out + (size_t)step * Bsz * S + (size_t)(rowBase + m) * S + half * 64;
    for (int s4 = 0; s4 < 16; ++s4) {
      float warr[4];
#pragma unroll
      for (int u = 0; u < 4; ++u) {
        int s = s4 * 4 + u;
        float w = big[m * VOCAB + half * 64 + s] * inv;
        warr[u] = w;
        const float4* er = (const float4*)(enc_row0 + s * H);
#pragma unroll
        for (int q4 = 0; q4 < 5; ++q4) {
          float4 v = er[q4];
          acc[q4 * 4 + 0] += w * v.x;
          acc[q4 * 4 + 1] += w * v.y;
          acc[q4 * 4 + 2] += w * v.z;
          acc[q4 * 4 + 3] += w * v.w;
        }
      }
      ((float4*)attn_row)[s4] = make_float4(warr[0], warr[1], warr[2], warr[3]);
    }
#pragma unroll
    for (int j = 0; j < 20; ++j) acc[j] += __shfl_xor(acc[j], 16, 32);
    if (half == 0) {
#pragma unroll
      for (int j = 0; j < 20; ++j) actx[m * HP + j] = acc[j];
    }
    __syncthreads();

    // comb = [d1, a] @ combW^T + combb  (tiny: VALU)
#pragma unroll
    for (int kk = 0; kk < 10; ++kk) {
      int k = half * 10 + kk;
      float cacc = combb[k];
      for (int j = 0; j < H; ++j) cacc += d1[m * HP + j] * combW[k * 2 * H + j];
      for (int j = 0; j < H; ++j)
        cacc += actx[m * HP + j] * combW[k * 2 * H + H + j];
      comb[m * HP + k] = cacc;
    }
    __syncthreads();

    // logits = comb @ outW^T + outb : 8 WMMA tiles of 16x16 -> LDS
    {
      v16h ca = a_frag(comb, HP, 0, H, lane);
#pragma unroll
      for (int nt = 0; nt < 8; ++nt) {
        v8f c = splat8(cbo[nt]);
        c = wmma16(ca, fout[nt], c);
        store_d(big, VOCAB, nt, c, lane);
      }
    }
    __syncthreads();

    // coalesced float4 store of the 16x128 logits tile
#pragma unroll
    for (int j = 0; j < 16; ++j) {
      int idx = lane + 32 * j;  // 0..511 float4s
      int row = idx >> 5, c4 = idx & 31;
      float4 v = ((const float4*)big)[row * 32 + c4];
      *(float4*)(vec_out + ((size_t)(rowBase + row) * MAXDEC + step) * VOCAB +
                 c4 * 4) = v;
    }

    // argmax over 128 logits (first-max semantics)
    float best = -3.4e38f;
    int bi = 0;
    for (int s = 0; s < 64; ++s) {
      int idx = half * 64 + s;
      float v = big[m * VOCAB + idx];
      if (v > best) { best = v; bi = idx; }
    }
    float ob = __shfl_xor(best, 16, 32);
    int oi = __shfl_xor(bi, 16, 32);
    if (ob > best || (ob == best && oi < bi)) { best = ob; bi = oi; }
    if (half == 0) vin[m] = bi;
    __syncthreads();
  }

  // hidden = stack([d0, d1])
#pragma unroll
  for (int kk = 0; kk < 10; ++kk) {
    int k = half * 10 + kk;
    hidden_out[(size_t)(rowBase + m) * H + k] = d0[m * HP + k];
    hidden_out[(size_t)Bsz * H + (size_t)(rowBase + m) * H + k] = d1[m * HP + k];
  }
}

extern "C" void kernel_launch(void* const* d_in, const int* in_sizes, int n_in,
                              void* d_out, int out_size, void* d_ws,
                              size_t ws_size, hipStream_t stream) {
  (void)in_sizes; (void)n_in; (void)out_size; (void)ws_size;
  const float* x = (const float*)d_in[0];
  const float* eWih0 = (const float*)d_in[1];
  const float* eWhh0 = (const float*)d_in[2];
  const float* ebih0 = (const float*)d_in[3];
  const float* ebhh0 = (const float*)d_in[4];
  const float* eWih1 = (const float*)d_in[5];
  const float* eWhh1 = (const float*)d_in[6];
  const float* ebih1 = (const float*)d_in[7];
  const float* ebhh1 = (const float*)d_in[8];
  const float* dWih0 = (const float*)d_in[9];
  const float* dWhh0 = (const float*)d_in[10];
  const float* dbih0 = (const float*)d_in[11];
  const float* dbhh0 = (const float*)d_in[12];
  const float* dWih1 = (const float*)d_in[13];
  const float* dWhh1 = (const float*)d_in[14];
  const float* dbih1 = (const float*)d_in[15];
  const float* dbhh1 = (const float*)d_in[16];
  const float* emb = (const float*)d_in[17];
  const float* qW = (const float*)d_in[18];
  const float* qb = (const float*)d_in[19];
  const float* combW = (const float*)d_in[20];
  const float* combb = (const float*)d_in[21];
  const float* outW = (const float*)d_in[22];
  const float* outb = (const float*)d_in[23];

  float* wsf = (float*)d_ws;
  float* encoded = wsf;                                // B*S*H floats (84 MB, L2-resident)
  float* h0 = encoded + (size_t)Bsz * S * H;
  float* h1 = h0 + (size_t)Bsz * H;

  float* out = (float*)d_out;
  float* vec_out = out;                                // [B,6,VOCAB]
  float* hidden = out + (size_t)Bsz * MAXDEC * VOCAB;  // [2,B,H]
  float* attn = hidden + (size_t)2 * Bsz * H;          // [6,B,S]

  dim3 blk(64);                  // 2 waves, one 16-row batch tile each
  dim3 grd(Bsz / 32);            // 256 blocks -> 512 independent recurrences

  t2s_encoder<<<grd, blk, 0, stream>>>(x, eWih0, eWhh0, ebih0, ebhh0, eWih1,
                                       eWhh1, ebih1, ebhh1, encoded, h0, h1);
  t2s_decoder<<<grd, blk, 0, stream>>>(
      dWih0, dWhh0, dbih0, dbhh0, dWih1, dWhh1, dbih1, dbhh1, emb, qW, qb,
      combW, combb, outW, outb, encoded, h0, h1, vec_out, hidden, attn);
}
